// HaloAttn_25091198943832
// MI455X (gfx1250) — matrix operand order repaired and checked by reading, not compiled, not run on hardware
//
#include <hip/hip_runtime.h>

typedef _Float16 f16;
typedef __attribute__((ext_vector_type(16))) _Float16 v16h;
typedef __attribute__((ext_vector_type(8)))  _Float16 v8h;
typedef __attribute__((ext_vector_type(8)))  float    v8f;

#define HEADS  8
#define DH     32
#define BS     8
#define HALO   3
#define WIN    14
#define NKEY   196      // 14*14
#define NKEYP  224      // padded to 7*32 for the P*V K-loop
#define RELN   27
#define CIN    256
#define IMG    128
#define NPIX   16384    // 128*128
#define SCALEF 0.17677669529663687f   // 1/sqrt(32)

static __device__ __forceinline__ v8f wmma16(v16h a, v16h b, v8f c) {
  // D = A(16x32 f16) * B(32x16 f16) + C(16x16 f32)
  return __builtin_amdgcn_wmma_f32_16x16x32_f16(false, a, false, b, (short)0, c,
                                                false, false);
}

// K-dim swizzle for A-side LDS tiles: swap element groups [8..15] and [16..23]
// so that the per-half A fragment (K = {8h..8h+7, 16+8h..16+8h+7}) is one
// contiguous 32-byte chunk -> single v16h LDS load, no VALU shuffles.
static __device__ __forceinline__ int swz(int d) {
  return (d & ~24) | ((d & 8) << 1) | ((d & 16) >> 1);
}

// ---------------------------------------------------------------------------
// Kernel 0: convert weights f32 -> f16
// ---------------------------------------------------------------------------
__global__ void cvt_weights(const float* __restrict__ wq,
                            const float* __restrict__ wkv,
                            f16* __restrict__ wq16, f16* __restrict__ wkv16) {
  int i = blockIdx.x * 256 + threadIdx.x;       // grid = 512 blocks -> 131072
  if (i < 256 * CIN) wq16[i] = (f16)wq[i];
  wkv16[i] = (f16)wkv[i];                       // 512*256 == 131072 exactly
}

// ---------------------------------------------------------------------------
// Kernel 1: projection GEMM. out[o][p] = sum_c w[o][c] * x[c][p]  (per batch)
// One WG (256 thr / 8 waves): 64 out-channels x 128 pixels; K-loop 256 by 32.
// Per K-step each wave issues 4 WMMAs sharing one B fragment.
// ---------------------------------------------------------------------------
__global__ __launch_bounds__(256) void proj_gemm(
    const float* __restrict__ x, const f16* __restrict__ wq16,
    const f16* __restrict__ wkv16, f16* __restrict__ q16,
    f16* __restrict__ kv16) {
  const int b  = blockIdx.z;
  const int o0 = blockIdx.x * 64;               // 12 tiles over 768 channels
  const int p0 = blockIdx.y * 128;              // pixel tile base
  const int t    = threadIdx.x;
  const int wv   = t >> 5;                      // wave 0..7 -> 16-pixel slice
  const int lane = t & 31;
  const int half = lane >> 4;
  const int m    = lane & 15;

  __shared__ alignas(64) f16 xsT[128][32];      // [pixel][chan]      8 KB
  __shared__ alignas(64) f16 wslP[64][32];      // [orow][chan]  4 KB

  const bool isq = (o0 < 256);                  // tiles 0..3 = q, 4..11 = kv
  const f16* wptr = isq ? (wq16 + (size_t)o0 * CIN)
                        : (wkv16 + (size_t)(o0 - 256) * CIN);
  f16* outp = isq ? (q16 + ((size_t)b * 256 + o0) * NPIX)
                  : (kv16 + ((size_t)b * 512 + (o0 - 256)) * NPIX);
  const float* xb = x + (size_t)b * CIN * NPIX;

  v8f acc[4] = {{}, {}, {}, {}};
  for (int k0 = 0; k0 < CIN; k0 += 32) {
    // stage x tile: 32 chan x 128 px, float4 global loads, f16 transposed
#pragma unroll
    for (int i = 0; i < 4; ++i) {
      int base = i * 1024 + t * 4;              // 0..4095, step 4
      int c = base >> 7;
      int p = base & 127;
      const float4 v = *(const float4*)&xb[(size_t)(k0 + c) * NPIX + p0 + p];
      xsT[p + 0][c] = (f16)v.x;
      xsT[p + 1][c] = (f16)v.y;
      xsT[p + 2][c] = (f16)v.z;
      xsT[p + 3][c] = (f16)v.w;
    }
    // stage weight tile 64x32 (f16 source), one v8h per thread, linear
    {
      int idx = t * 8;                          // 0..2047
      int r = idx >> 5, c = idx & 31;           // c multiple of 8
      v8h wv8 = *(const v8h*)&wptr[(size_t)r * CIN + k0 + c];
      *(v8h*)&wslP[r][c] = wv8;
    }
    __syncthreads();

    const int pn = wv * 16 + m;
    v16h bf = *(const v16h*)&xsT[pn][16 * half];
#pragma unroll
    for (int at = 0; at < 4; ++at) {
      v16h a = *(const v16h*)&wslP[16 * at + m][16 * half];
      acc[at] = wmma16(a, bf, acc[at]);
    }
    __syncthreads();
  }

  // D layout: VGPR r -> row M = r + 8*half, col N = m
  const int pn = wv * 16 + m;
#pragma unroll
  for (int at = 0; at < 4; ++at)
#pragma unroll
    for (int r = 0; r < 8; ++r)
      outp[(size_t)(16 * at + r + 8 * half) * NPIX + p0 + pn] =
          (f16)acc[at][r];
}

// ---------------------------------------------------------------------------
// Kernel 2: fused halo attention for one (batch*head, 8x8 block) tile.
// 128 threads = 4 waves; wave wq owns query tile [16*wq, 16*wq+16).
// ---------------------------------------------------------------------------
__global__ __launch_bounds__(128) void halo_attn(
    const f16* __restrict__ q16, const f16* __restrict__ kv16,
    const float* __restrict__ hrel, const float* __restrict__ wrel,
    float* __restrict__ out) {
  const int n  = blockIdx.x;                    // block 0..255
  const int bh = blockIdx.y;                    // 0..31
  const int b = bh >> 3, h = bh & 7;
  const int by = n >> 4, bx = n & 15;
  const int t = threadIdx.x;
  const int wq = t >> 5;                        // wave / query tile
  const int lane = t & 31;
  const int half = lane >> 4;
  const int m = lane & 15;

  __shared__ alignas(64) f16 qs[64][32];        //  4096 B  Q, linear K-dim
  __shared__ alignas(64) f16 Rh16[64][32];      //  4096 B  rel-height logits
  __shared__ alignas(64) f16 Rw16[64][32];      //  4096 B  rel-width logits
  __shared__ alignas(64) f16 relhL[RELN][32];   //  1728 B  height_rel (f16)
  __shared__ alignas(64) f16 relwL[RELN][32];   //  1728 B  width_rel  (f16)
  __shared__ alignas(64) f16 vsT[32][NKEYP];    // 14336 B  V transposed
  __shared__ alignas(64) f16 kps[64 * NKEYP];   // 28672 B  K (ph.1)/P (ph.2)
  f16 (*ks)[32]    = (f16(*)[32])kps;           // [224][32]
  f16 (*ps)[NKEYP] = (f16(*)[NKEYP])kps;        // [64][224], swizzled chunks

  const f16* qb = q16 + ((size_t)b * 256 + 32 * h) * NPIX;
  const f16* kb = kv16 + ((size_t)b * 512 + 64 * h) * NPIX;
  const f16* vb = kb + (size_t)32 * NPIX;

  // ---- stage Q (v8h row loads, linear K-dim scatter) ----
#pragma unroll
  for (int i = 0; i < 2; ++i) {
    int lin = (i * 128 + t) * 8;                // 0..2040, step 8
    int d = lin >> 6;                           // 0..31
    int q0 = lin & 63;                          // multiple of 8 -> one row
    int y = by * 8 + (q0 >> 3);
    v8h qv = *(const v8h*)&qb[(size_t)d * NPIX + y * IMG + bx * 8];
    int sd = d;
#pragma unroll
    for (int j = 0; j < 8; ++j) qs[q0 + j][sd] = qv[j];
  }
  // ---- stage rel tables (f16, linear: B-side) ----
  for (int i = 0; i < 7; ++i) {
    int idx = i * 128 + t;
    if (idx < RELN * DH) {
      relhL[idx >> 5][idx & 31] = (f16)hrel[idx];
      relwL[idx >> 5][idx & 31] = (f16)wrel[idx];
    }
  }
  // ---- stage K and V^T, zero-padding OOB halo and keys >= 196 ----
  for (int i = 0; i < 56; ++i) {
    int idx = i * 128 + t;                      // 7168 = 224wp * 32d
    int d = idx / NKEYP;
    int wp = idx - d * NKEYP;
    f16 kvv = (f16)0.f, vvv = (f16)0.f;
    if (wp < NKEY) {
      int wi = wp / WIN, wj = wp - wi * WIN;
      int y = by * 8 + wi - HALO, xx = bx * 8 + wj - HALO;
      if (y >= 0 && y < IMG && xx >= 0 && xx < IMG) {
        size_t off = (size_t)d * NPIX + y * IMG + xx;
        kvv = kb[off];
        vvv = vb[off];
      }
    }
    ks[wp][d]  = kvv;
    vsT[d][swz(wp)] = vvv;
  }
  __syncthreads();

  // ---- A fragment: this wave's 16 queries (single v16h load) ----
  const int qrow = 16 * wq + m;
  v16h aq = *(const v16h*)&qs[qrow][16 * half];

  // ---- relative-position logits: R = Q * rel^T (27 cols, 2 N-tiles) ----
#pragma unroll
  for (int rt = 0; rt < 2; ++rt) {
    const f16 (*relL)[32] = rt ? relwL : relhL;
    f16* Rout = rt ? &Rw16[0][0] : &Rh16[0][0];
#pragma unroll
    for (int ntl = 0; ntl < 2; ++ntl) {
      int ncol = ntl * 16 + m;
      int nrow = ncol < RELN ? ncol : (RELN - 1);   // clamp (cols>=27 unused)
      v16h bf = *(const v16h*)&relL[nrow][16 * half];
      v8f c = {};
      c = wmma16(aq, bf, c);
#pragma unroll
      for (int r = 0; r < 8; ++r)
        Rout[(16 * wq + r + 8 * half) * 32 + ncol] = (f16)c[r];
    }
  }

  // ---- S = Q * K^T : 13 key tiles of 16, single K=32 WMMA each ----
  v8f S[13];
#pragma unroll
  for (int kt = 0; kt < 13; ++kt) {
    v16h bk = *(const v16h*)&ks[kt * 16 + m][16 * half];
    v8f c = {};
    S[kt] = wmma16(aq, bk, c);
  }
  __syncthreads();   // all waves done reading ks before it is reused as ps

  // ---- softmax over 196 real keys (cols >= 196 masked to -inf) ----
  int kkA[13], riA[13], rjA[13], colP[13];
#pragma unroll
  for (int kt = 0; kt < 13; ++kt) {
    int kk = kt * 16 + m;
    kkA[kt] = kk;
    riA[kt] = kk / WIN;
    rjA[kt] = kk - riA[kt] * WIN;
    colP[kt] = 32 * (kt >> 1) + swz(((kt & 1) << 4) + m);  // swizzled P col
  }
#pragma unroll
  for (int r = 0; r < 8; ++r) {
    const int qq = 16 * wq + r + 8 * half;      // D row for this VGPR/half
    const int hq = qq >> 3, wc = qq & 7;
    float mrow = -3.0e38f;
#pragma unroll
    for (int kt = 0; kt < 13; ++kt) {
      float v = -3.0e38f;
      if (kkA[kt] < NKEY)
        v = S[kt][r] * SCALEF + (float)Rh16[qq][riA[kt] - hq + 13]
                              + (float)Rw16[qq][rjA[kt] - wc + 13];
      S[kt][r] = v;
      mrow = fmaxf(mrow, v);
    }
#pragma unroll
    for (int s = 1; s < 16; s <<= 1) mrow = fmaxf(mrow, __shfl_xor(mrow, s, 32));
    float sum = 0.f;
#pragma unroll
    for (int kt = 0; kt < 13; ++kt) {
      float e = __expf(S[kt][r] - mrow);
      S[kt][r] = e;
      sum += e;
    }
#pragma unroll
    for (int s = 1; s < 16; s <<= 1) sum += __shfl_xor(sum, s, 32);
    const float inv = 1.f / sum;
#pragma unroll
    for (int kt = 0; kt < 13; ++kt)
      ps[qq][colP[kt]] = (f16)(S[kt][r] * inv);
  }
  // zero P keys 208..223 (physical positions after swizzle)
  {
    const int qq = 16 * wq + m;
#pragma unroll
    for (int e = 0; e < 8; ++e)
      ps[qq][192 + swz(16 + 8 * half + e)] = (f16)0.f;
  }
  __syncthreads();

  // ---- O = P(64x224) * V(224x32): 7 K-steps x 2 N-tiles ----
  v8f O0 = {}, O1 = {};
#pragma unroll
  for (int s = 0; s < 7; ++s) {
    v16h ap = *(const v16h*)&ps[qrow][32 * s + 16 * half];
    v16h b0 = *(const v16h*)&vsT[m][32 * s + 16 * half];
    v16h b1 = *(const v16h*)&vsT[m + 16][32 * s + 16 * half];
    O0 = wmma16(ap, b0, O0);
    O1 = wmma16(ap, b1, O1);
  }

  // ---- write out[b][32h + d][y][x] (f32) ----
  float* ob = out + ((size_t)b * 256 + 32 * h) * NPIX;
#pragma unroll
  for (int r = 0; r < 8; ++r) {
    int qq = 16 * wq + r + 8 * half;
    int y = by * 8 + (qq >> 3), xx = bx * 8 + (qq & 7);
    ob[(size_t)m * NPIX + y * IMG + xx]        = O0[r];
    ob[(size_t)(m + 16) * NPIX + y * IMG + xx] = O1[r];
  }
}

// ---------------------------------------------------------------------------
extern "C" void kernel_launch(void* const* d_in, const int* in_sizes, int n_in,
                              void* d_out, int out_size, void* d_ws,
                              size_t ws_size, hipStream_t stream) {
  (void)in_sizes; (void)n_in; (void)out_size; (void)ws_size;
  const float* x    = (const float*)d_in[0];
  const float* w_q  = (const float*)d_in[1];
  const float* w_kv = (const float*)d_in[2];
  const float* hrel = (const float*)d_in[3];
  const float* wrel = (const float*)d_in[4];
  float* out = (float*)d_out;

  char* ws = (char*)d_ws;
  f16* wq16  = (f16*)(ws);                                   // 131072 B
  f16* wkv16 = (f16*)(ws + 131072);                          // 262144 B
  f16* q16   = (f16*)(ws + 131072 + 262144);                 // 32 MB
  f16* kv16  = (f16*)(ws + 131072 + 262144 + 33554432);      // 64 MB

  cvt_weights<<<512, 256, 0, stream>>>(w_q, w_kv, wq16, wkv16);

  dim3 g1(12, 128, 4);       // 768/64 out-channel tiles, 16384/128 px, 4 batch
  proj_gemm<<<g1, 256, 0, stream>>>(x, wq16, wkv16, q16, kv16);

  dim3 g2(256, 32);          // 256 blocks, 4 batches * 8 heads
  halo_attn<<<g2, 128, 0, stream>>>(q16, kv16, hrel, wrel, out);
}
